// PyramidCompressor_60344290509593
// MI455X (gfx1250) — matrix operand order repaired and checked by reading, not compiled, not run on hardware
//
#include <hip/hip_runtime.h>
#include <stdint.h>

typedef __bf16 bf16_t;
typedef __attribute__((ext_vector_type(16))) __bf16 v16bf;
typedef __attribute__((ext_vector_type(8)))  __bf16 v8bf;
typedef __attribute__((ext_vector_type(8)))  float  v8f;
typedef __attribute__((ext_vector_type(4)))  float  v4f;

// TDM descriptor SGPR groups
typedef __attribute__((ext_vector_type(4))) unsigned int u32x4;
typedef __attribute__((ext_vector_type(8))) int          i32x8;
typedef __attribute__((ext_vector_type(4))) int          i32x4;

#define LDA 40   // 32 + 8 pad elements: 64B row + 16B pad (matches TDM pad cfg)
#define LDB 40

union Frag { v16bf v; v8bf h[2]; };

// ---------------------------------------------------------------------------
// Rank kernel: rank[i] = #{j : imp[j] > imp[i]  ||  (imp[j]==imp[i] && j<i)}
// == stable argsort of -importance. Writes sorted_idx[rank] = i.
// ---------------------------------------------------------------------------
__global__ __launch_bounds__(256)
void rank_kernel(const float* __restrict__ imp, int* __restrict__ sidx, int n) {
  __shared__ float tile[1024];
  int gid = blockIdx.x * 256 + threadIdx.x;
  float my = imp[gid];
  int rank = 0;
  for (int t0 = 0; t0 < n; t0 += 1024) {
    for (int i = threadIdx.x; i < 1024; i += 256) tile[i] = imp[t0 + i];
    __syncthreads();
    #pragma unroll 4
    for (int j = 0; j < 1024; ++j) {
      float o = tile[j];
      rank += (o > my) || (o == my && (t0 + j) < gid);
    }
    __syncthreads();
  }
  sidx[rank] = gid;
}

// ---------------------------------------------------------------------------
// Level 0: identity copy of the top-L0 tokens (k and v halves of d_out).
// ---------------------------------------------------------------------------
__global__ __launch_bounds__(256)
void copy_level0(const float* __restrict__ gK, const float* __restrict__ gV,
                 const int* __restrict__ sidx, float* __restrict__ out, long vOff) {
  int tok = sidx[blockIdx.x];
  const v4f* sk = (const v4f*)(gK + (size_t)tok * 2048);
  const v4f* sv = (const v4f*)(gV + (size_t)tok * 2048);
  v4f* dk = (v4f*)(out + (size_t)tok * 2048);
  v4f* dv = (v4f*)(out + (size_t)vOff + (size_t)tok * 2048);
  for (int i = threadIdx.x; i < 512; i += 256) { dk[i] = sk[i]; dv[i] = sv[i]; }
}

// ---------------------------------------------------------------------------
// f32 -> bf16 weight conversion with zero padding to (Kp x Np).
// ---------------------------------------------------------------------------
__global__ __launch_bounds__(256)
void convert_w(const float* __restrict__ W, bf16_t* __restrict__ dst,
               int K, int N, int Kp, int Np) {
  long total = (long)Kp * Np;
  for (long i = (long)blockIdx.x * blockDim.x + threadIdx.x; i < total;
       i += (long)gridDim.x * blockDim.x) {
    int  n = (int)(i % Np);
    long k = i / Np;
    float v = (k < K && n < N) ? W[k * (long)N + n] : 0.0f;
    dst[i] = (bf16_t)v;
  }
}

// ---------------------------------------------------------------------------
// Generic relu(A @ W + b) GEMM on 128x128 tiles, K-step 32, bf16 WMMA, f32 acc.
//   a_mode   : 0 = gather f32 rows (keys rows [0,halfM), values rows [halfM,M))
//              1 = dense bf16 intermediate (lda leading dim) -> A tile fetched
//                  by the Tensor Data Mover (tensor_load_to_lds, TENSORcnt)
//   out_mode : 0 = dense bf16 intermediate (ldo leading dim, zeros in pad cols)
//              1 = f32 scatter into d_out via sorted_idx (k half / v half)
// N is the padded column count (multiple of 32); realN bounds bias/valid cols.
// K must be a multiple of 32 (all layers: 2048 / 1024 / 416).
// ---------------------------------------------------------------------------
__global__ __launch_bounds__(256)
void gemm_bf16_wmma(int a_mode,
                    const float* __restrict__ gK, const float* __restrict__ gV,
                    const int* __restrict__ sidx, int idx_start, int halfM,
                    const bf16_t* __restrict__ Adense, int lda,
                    const bf16_t* __restrict__ W,
                    const float* __restrict__ bias, int realN,
                    int M, int N, int K,
                    int out_mode, bf16_t* __restrict__ Obf, int ldo,
                    float* __restrict__ OutF, long vOff) {
  __shared__ __align__(16) bf16_t Alds[128 * LDA];
  __shared__ __align__(16) bf16_t Blds[128 * LDB];

  const int tid  = threadIdx.x;
  const int wave = tid >> 5;     // 8 waves; wave w owns tile-row w (16 rows)
  const int lane = tid & 31;
  const int kh   = lane >> 4;    // which K-half of the fragment this lane holds
  const int lm   = lane & 15;    // row (A) / col (B,D) within 16x16 tile

  const int m0 = blockIdx.y * 128;
  const int n0 = blockIdx.x * 128;

  v8f acc[8];
  #pragma unroll
  for (int c = 0; c < 8; ++c)
    #pragma unroll
    for (int r = 0; r < 8; ++r) acc[c][r] = 0.0f;

  const int ar = tid >> 1;       // A-tile row this thread fills (0..127)
  const int ah = tid & 1;        // which 16-element K chunk

  // ---- TDM descriptor pieces that are loop-invariant (dense-A mode) ----
  const unsigned ldsA  = (unsigned)(uintptr_t)(void*)Alds;   // LDS byte offset
  const unsigned rowsv = (unsigned)(M - m0);                 // valid rows (<64K)

  for (int k0 = 0; k0 < K; k0 += 32) {
    if (a_mode == 1) {
      // ---- Tensor Data Mover: DMA A tile (128 rows x 64B, stride lda) ----
      if (wave == 0) {
        unsigned long long ga =
            (unsigned long long)(uintptr_t)(Adense + (size_t)m0 * lda + k0);
        u32x4 g0;
        g0[0] = 1u;                                   // count=1, user D#
        g0[1] = ldsA;                                 // lds_addr
        g0[2] = (unsigned)ga;                         // global_addr[31:0]
        g0[3] = (unsigned)((ga >> 32) & 0x01FFFFFFu)  // global_addr[56:32]
                | (2u << 30);                         // type = 2 ("image")
        i32x8 g1;
        g1[0] = (int)((1u << 16)      // data_size = 2 bytes
                    | (1u << 20)      // pad_enable
                    | (3u << 22)      // pad_interval: 16 DWORDs (64B rows)
                    | (3u << 25));    // pad_amount: 4 DWORDs (16B -> LDA=40)
        g1[1] = (int)(32u << 16);                     // tensor_dim0 = 32
        g1[2] = (int)((rowsv & 0xFFFFu) << 16);       // tensor_dim1 lo (OOB=0)
        g1[3] = (int)((rowsv >> 16) | (32u << 16));   // dim1 hi | tile_dim0=32
        g1[4] = 128;                                  // tile_dim1=128, dim2=0
        g1[5] = lda;                                  // tensor_dim0_stride lo
        g1[6] = 0;                                    // stride hi, dim1_stride
        g1[7] = 0;
        i32x4 gz = {0, 0, 0, 0};                      // 2D: groups 2/3 unused
#if defined(__clang_major__) && (__clang_major__ >= 23)
        i32x8 gz8 = {0, 0, 0, 0, 0, 0, 0, 0};
        __builtin_amdgcn_tensor_load_to_lds(g0, g1, gz, gz, gz8, 0);
#else
        __builtin_amdgcn_tensor_load_to_lds(g0, g1, gz, gz, 0);
#endif
      }
    } else {
      // ---- gather-mode A tile: f32 rows via sorted_idx, convert to bf16 ----
      bf16_t* dst = &Alds[ar * LDA + ah * 16];
      int m = m0 + ar;
      if (m < M) {
        int mm  = (m < halfM) ? m : m - halfM;
        int tok = sidx[idx_start + mm];
        const float* src =
            ((m < halfM) ? gK : gV) + (size_t)tok * 2048 + k0 + ah * 16;
        #pragma unroll
        for (int q = 0; q < 4; ++q) {
          v4f f = *(const v4f*)(src + q * 4);
          dst[q * 4 + 0] = (bf16_t)f.x;
          dst[q * 4 + 1] = (bf16_t)f.y;
          dst[q * 4 + 2] = (bf16_t)f.z;
          dst[q * 4 + 3] = (bf16_t)f.w;
        }
      } else {
        #pragma unroll
        for (int q = 0; q < 16; ++q) dst[q] = (bf16_t)0.0f;
      }
    }

    // ---- stage B tile (32 x 128) transposed into LDS as [n][k] ----
    #pragma unroll
    for (int i = 0; i < 16; ++i) {
      int e  = i * 256 + tid;     // 0..4095
      int kk = e >> 7;            // 0..31
      int nn = e & 127;
      int gn = n0 + nn;
      bf16_t v = (bf16_t)0.0f;
      if (gn < N) v = W[(size_t)(k0 + kk) * (size_t)N + gn];
      Blds[nn * LDB + kk] = v;
    }

    // TDM completion (wave 0) must precede the barrier everyone waits on.
    if (a_mode == 1 && wave == 0) __builtin_amdgcn_s_wait_tensorcnt(0);
    __syncthreads();

    // ---- A fragment: row M=lm, K runs [kh*8, kh*8+8) and [16+kh*8, ...) ----
    Frag af;
    {
      const bf16_t* ap = &Alds[(wave * 16 + lm) * LDA + kh * 8];
      af.h[0] = *(const v8bf*)ap;
      af.h[1] = *(const v8bf*)(ap + 16);
    }
    // ---- 8 WMMAs: one per 16-wide output tile in this wave's strip ----
    #pragma unroll
    for (int c = 0; c < 8; ++c) {
      Frag bfr;
      const bf16_t* bp = &Blds[(c * 16 + lm) * LDB + kh * 8];
      bfr.h[0] = *(const v8bf*)bp;
      bfr.h[1] = *(const v8bf*)(bp + 16);
      acc[c] = __builtin_amdgcn_wmma_f32_16x16x32_bf16(
          false, af.v, false, bfr.v, (short)0, acc[c], false, false);
    }
    __syncthreads();
  }

  // ---- bias + relu + store (D layout: VGPR r -> M = 8*kh + r, lane lm -> N) --
  #pragma unroll
  for (int c = 0; c < 8; ++c) {
    int col = n0 + c * 16 + lm;
    if (col >= N) continue;
    float b = (col < realN) ? bias[col] : 0.0f;
    int rowbase = m0 + wave * 16 + kh * 8;
    #pragma unroll
    for (int r = 0; r < 8; ++r) {
      int row = rowbase + r;
      if (row >= M) continue;
      float v = fmaxf(acc[c][r] + b, 0.0f);
      if (out_mode == 0) {
        Obf[(size_t)row * ldo + col] = (col < realN) ? (bf16_t)v : (bf16_t)0.0f;
      } else {
        int mm  = (row < halfM) ? row : row - halfM;
        int tok = sidx[idx_start + mm];
        size_t base = (row < halfM) ? 0 : (size_t)vOff;
        OutF[base + (size_t)tok * 2048 + col] = v;
      }
    }
  }
}

// ---------------------------------------------------------------------------
extern "C" void kernel_launch(void* const* d_in, const int* in_sizes, int n_in,
                              void* d_out, int out_size, void* d_ws, size_t ws_size,
                              hipStream_t stream) {
  const float* keys   = (const float*)d_in[0];
  const float* values = (const float*)d_in[1];
  const float* imp    = (const float*)d_in[2];
  const float* Wenc0  = (const float*)d_in[3];
  const float* benc0  = (const float*)d_in[4];
  const float* Wenc1  = (const float*)d_in[5];
  const float* benc1  = (const float*)d_in[6];
  const float* Wdec0  = (const float*)d_in[7];
  const float* bdec0  = (const float*)d_in[8];
  const float* Wdec1  = (const float*)d_in[9];
  const float* bdec1  = (const float*)d_in[10];
  float* out = (float*)d_out;

  const int NTOK = 16384, H = 2048, D0 = 1024, D1 = 409, D1P = 416;
  const int L0 = 3276, L1 = 2097, L2 = 11011;    // level sizes (from reference)
  const int S1 = 3276, S2 = 5373;                // sorted_idx starts for lvl1/2
  const long VOFF = (long)NTOK * H;              // offset of out_v in d_out

  // workspace layout (256B aligned); total ~74 MB
  char* p = (char*)d_ws;
  auto take = [&](size_t bytes) {
    char* r = p; p += (bytes + 255) & ~(size_t)255; return r;
  };
  int*    sidx   = (int*)   take(sizeof(int) * NTOK);
  bf16_t* wenc0b = (bf16_t*)take((size_t)H   * D0  * 2);
  bf16_t* wdec0b = (bf16_t*)take((size_t)D0  * H   * 2);
  bf16_t* wenc1b = (bf16_t*)take((size_t)D0  * D1P * 2);
  bf16_t* wdec1b = (bf16_t*)take((size_t)D1P * D0  * 2);
  bf16_t* bufA   = (bf16_t*)take((size_t)(2 * L2) * D0  * 2); // T1 / T2a / T2c
  bf16_t* bufB   = (bf16_t*)take((size_t)(2 * L2) * D1P * 2); // T2b

  rank_kernel<<<NTOK / 256, 256, 0, stream>>>(imp, sidx, NTOK);

  convert_w<<<2048, 256, 0, stream>>>(Wenc0, wenc0b, H,  D0, H,   D0);
  convert_w<<<2048, 256, 0, stream>>>(Wdec0, wdec0b, D0, H,  D0,  H);
  convert_w<<<1024, 256, 0, stream>>>(Wenc1, wenc1b, D0, D1, D0,  D1P);
  convert_w<<<1024, 256, 0, stream>>>(Wdec1, wdec1b, D1, D0, D1P, D0);

  copy_level0<<<L0, 256, 0, stream>>>(keys, values, sidx, out, VOFF);

  // ---- Level 1: enc0 -> dec0 on 2*L1 rows ----
  {
    int M = 2 * L1;
    dim3 g1((D0 + 127) / 128, (M + 127) / 128);
    gemm_bf16_wmma<<<g1, 256, 0, stream>>>(0, keys, values, sidx, S1, L1,
        nullptr, 0, wenc0b, benc0, D0, M, D0, H, 0, bufA, D0, out, VOFF);
    dim3 g2((H + 127) / 128, (M + 127) / 128);
    gemm_bf16_wmma<<<g2, 256, 0, stream>>>(1, keys, values, sidx, S1, L1,
        bufA, D0, wdec0b, bdec0, H, M, H, D0, 1, nullptr, 0, out, VOFF);
  }
  // ---- Level 2: enc0 -> enc1 -> dec1 -> dec0 on 2*L2 rows ----
  {
    int M = 2 * L2;
    dim3 g3((D0 + 127) / 128, (M + 127) / 128);
    gemm_bf16_wmma<<<g3, 256, 0, stream>>>(0, keys, values, sidx, S2, L2,
        nullptr, 0, wenc0b, benc0, D0, M, D0, H, 0, bufA, D0, out, VOFF);
    dim3 g4((D1P + 127) / 128, (M + 127) / 128);
    gemm_bf16_wmma<<<g4, 256, 0, stream>>>(1, keys, values, sidx, S2, L2,
        bufA, D0, wenc1b, benc1, D1, M, D1P, D0, 0, bufB, D1P, out, VOFF);
    dim3 g5((D0 + 127) / 128, (M + 127) / 128);
    gemm_bf16_wmma<<<g5, 256, 0, stream>>>(1, keys, values, sidx, S2, L2,
        bufB, D1P, wdec1b, bdec1, D0, M, D0, D1P, 0, bufA, D0, out, VOFF);
    dim3 g6((H + 127) / 128, (M + 127) / 128);
    gemm_bf16_wmma<<<g6, 256, 0, stream>>>(1, keys, values, sidx, S2, L2,
        bufA, D0, wdec0b, bdec0, H, M, H, D0, 1, nullptr, 0, out, VOFF);
  }
  (void)in_sizes; (void)n_in; (void)out_size; (void)ws_size;
}
